// EncoderDecoderNet_35390530519329
// MI455X (gfx1250) — matrix operand order repaired and checked by reading, not compiled, not run on hardware
//
#include <hip/hip_runtime.h>
#include <hip/hip_bf16.h>

#define NQ 100000
#define NL 32
#define QD 384
#define LD 1024
#define HD 128
#define E_ALL 1000000
#define E_SEE 750000
#define E_PRED 250000
#define NN (NQ + NL)
#define BN_EPS 1e-5f
#define LDS_PITCH 34

typedef __bf16 bf16_t;
typedef __attribute__((ext_vector_type(16))) __bf16 v16bf;
typedef __attribute__((ext_vector_type(8)))  float  v8f;

// ---------------------------------------------------------------------------
// WMMA GEMM: out[16 rows x 128 cols per block] = A[rows x K] @ W[K x 128] (+epilogue)
// 256 threads = 8 waves; wave w owns cols 16w..16w+15. A staged in LDS as bf16.
// MODE 0: out = acc + bias0[col]
// MODE 1: out = acc + deg[row]*(bias0[col]+bias1[col]) + wsum[row]*We[col] + res[row,col]
// ---------------------------------------------------------------------------
template <int MODE>
__global__ __launch_bounds__(256)
void gemm_kernel(const float* __restrict__ A, int K,
                 const float* __restrict__ W,
                 const float* __restrict__ bias0,
                 const float* __restrict__ bias1,
                 const float* __restrict__ We,
                 const float* __restrict__ degv,
                 const float* __restrict__ wsumv,
                 const float* __restrict__ res,
                 float* __restrict__ out)
{
    __shared__ bf16_t As[16 * LDS_PITCH];
    const int row0 = blockIdx.x * 16;
    const int tid  = threadIdx.x;
    const int lane = tid & 31;
    const int wave = tid >> 5;
    const int col0 = wave * 16;

    // A-matrix 16x32 bf16 lane mapping (ISA 7.12.2)
    const int mA  = lane & 15;
    const int kbA = (lane & 16) ? 8 : 0;
    // B-matrix 32x16 bf16 lane mapping
    const int nB  = lane & 15;
    const int col = col0 + nB;
    const int kbB = (lane & 16) ? 8 : 0;

    v8f acc = {};
    for (int kk = 0; kk < K; kk += 32) {
        __syncthreads();   // previous iteration's LDS reads complete
        {
            int m0 = tid >> 5, k0 = tid & 31;                // elems 0..255
            As[m0 * LDS_PITCH + k0] =
                (bf16_t)A[(size_t)(row0 + m0) * K + kk + k0];
            int e1 = tid + 256;                              // elems 256..511
            int m1 = e1 >> 5, k1 = e1 & 31;
            As[m1 * LDS_PITCH + k1] =
                (bf16_t)A[(size_t)(row0 + m1) * K + kk + k1];
        }
        __syncthreads();

        v16bf a, b;
#pragma unroll
        for (int v = 0; v < 8; ++v) {
            int ka = kbA + ((v < 4) ? (2 * v) : (16 + 2 * (v - 4)));
            a[2 * v]     = As[mA * LDS_PITCH + ka];
            a[2 * v + 1] = As[mA * LDS_PITCH + ka + 1];
        }
#pragma unroll
        for (int v = 0; v < 8; ++v) {
            int kb = kk + kbB + ((v < 4) ? (2 * v) : (16 + 2 * (v - 4)));
            b[2 * v]     = (bf16_t)W[(size_t)kb * HD + col];
            b[2 * v + 1] = (bf16_t)W[(size_t)(kb + 1) * HD + col];
        }
        acc = __builtin_amdgcn_wmma_f32_16x16x32_bf16(
            false, a, false, b, (short)0, acc, false, false);
    }

    const int mbase = (lane & 16) ? 8 : 0;   // D: VGPR r -> row r (lanes 0-15) / r+8
    if (MODE == 0) {
        float bc = bias0[col];
#pragma unroll
        for (int r = 0; r < 8; ++r) {
            int g = row0 + mbase + r;
            out[(size_t)g * HD + col] = acc[r] + bc;
        }
    } else {
        float bsum = bias0[col] + bias1[col];
        float wec  = We[col];
#pragma unroll
        for (int r = 0; r < 8; ++r) {
            int g = row0 + mbase + r;
            out[(size_t)g * HD + col] =
                acc[r] + degv[g] * bsum + wsumv[g] * wec + res[(size_t)g * HD + col];
        }
    }
}

// ---------------------------------------------------------------------------
// Edge prep: compact src/dst for can-see edges; accumulate wsum (edge gate) & deg
// ---------------------------------------------------------------------------
__global__ void edge_prep_kernel(const int* __restrict__ ei,
                                 const int* __restrict__ can_see,
                                 const float* __restrict__ eweight,
                                 const float* __restrict__ W_em,
                                 const float* __restrict__ b_em,
                                 int* __restrict__ srcc, int* __restrict__ dstc,
                                 float* __restrict__ wsum, float* __restrict__ degv)
{
    int i = blockIdx.x * blockDim.x + threadIdx.x;
    if (i >= E_SEE) return;
    int p = can_see[i];
    int s = ei[p];
    int d = ei[E_ALL + p];
    srcc[i] = s;
    dstc[i] = d;
    float w = eweight[p] * W_em[0] + b_em[0];
    w = (w > 0.f) ? w : 0.01f * w;           // leaky_relu, slope 0.01
    atomicAdd(&wsum[d], w);
    atomicAdd(&degv[d], 1.0f);
}

// ---------------------------------------------------------------------------
// Scatter-add: S[dst, c] += x[src, c]   (thread = (edge, channel), coalesced;
// S (51 MB) lives in the 192 MB L2, atomics resolve there)
// ---------------------------------------------------------------------------
__global__ void scatter_kernel(const int* __restrict__ srcc,
                               const int* __restrict__ dstc,
                               const float* __restrict__ x,
                               float* __restrict__ S)
{
    unsigned idx = blockIdx.x * blockDim.x + threadIdx.x;
    if (idx >= (unsigned)E_SEE * HD) return;
    int e = idx >> 7;
    int c = idx & 127;
    atomicAdd(&S[(size_t)dstc[e] * HD + c], x[(size_t)srcc[e] * HD + c]);
}

// ---------------------------------------------------------------------------
// BatchNorm stats: per-channel sum & sumsq (thread = channel, coalesced rows)
// ---------------------------------------------------------------------------
__global__ __launch_bounds__(128)
void bn_stats_kernel(const float* __restrict__ x, float* __restrict__ stats,
                     int rows_per_block)
{
    int c  = threadIdx.x;
    int r0 = blockIdx.x * rows_per_block;
    int r1 = r0 + rows_per_block;
    if (r1 > NN) r1 = NN;
    float s = 0.f, sq = 0.f;
    for (int r = r0; r < r1; ++r) {
        float v = x[(size_t)r * HD + c];
        s += v;
        sq += v * v;
    }
    atomicAdd(&stats[c], s);
    atomicAdd(&stats[HD + c], sq);
}

__global__ void bn_apply_kernel(float* __restrict__ x,
                                const float* __restrict__ stats,
                                const float* __restrict__ g,
                                const float* __restrict__ beta,
                                int do_lrelu)
{
    unsigned idx = blockIdx.x * blockDim.x + threadIdx.x;
    if (idx >= (unsigned)NN * HD) return;
    int c = idx & 127;
    const float invN = 1.0f / (float)NN;
    float mu  = stats[c] * invN;
    float var = stats[HD + c] * invN - mu * mu;   // biased var (jnp default)
    float y = g[c] * (x[idx] - mu) * rsqrtf(var + BN_EPS) + beta[c];
    if (do_lrelu) y = (y > 0.f) ? y : 0.01f * y;
    x[idx] = y;
}

// ---------------------------------------------------------------------------
// Edge prediction: one wave per edge, 128-dim dot via float4 + wave reduction
// ---------------------------------------------------------------------------
__global__ __launch_bounds__(256)
void edge_predict_kernel(const float* __restrict__ x_ini,
                         const float* __restrict__ x2,
                         const int* __restrict__ ei,
                         const int* __restrict__ emask,
                         float* __restrict__ out)
{
    int e = blockIdx.x * 8 + (threadIdx.x >> 5);
    int lane = threadIdx.x & 31;
    if (e >= E_PRED) return;
    int p = emask[e];
    int s = ei[p];
    int d = ei[E_ALL + p];
    const float4* xs = (const float4*)(x_ini + (size_t)s * HD);
    const float4* xd = (const float4*)(x2 + (size_t)d * HD);
    float4 a = xs[lane], b = xd[lane];
    float sum = a.x * b.x + a.y * b.y + a.z * b.z + a.w * b.w;
#pragma unroll
    for (int off = 16; off > 0; off >>= 1)
        sum += __shfl_down(sum, off, 32);
    if (lane == 0)
        out[e] = 1.0f / (1.0f + expf(-sum * (1.0f / (float)HD)));
}

// ---------------------------------------------------------------------------
extern "C" void kernel_launch(void* const* d_in, const int* in_sizes, int n_in,
                              void* d_out, int out_size, void* d_ws, size_t ws_size,
                              hipStream_t stream)
{
    const float* qf      = (const float*)d_in[0];
    const float* lf      = (const float*)d_in[1];
    const int*   ei      = (const int*)d_in[2];
    const int*   emask   = (const int*)d_in[3];
    const int*   ecansee = (const int*)d_in[4];
    const float* eweight = (const float*)d_in[5];
    const float* Wq   = (const float*)d_in[6];
    const float* bq   = (const float*)d_in[7];
    const float* Wl   = (const float*)d_in[8];
    const float* bl   = (const float*)d_in[9];
    const float* W_em = (const float*)d_in[10];
    const float* b_em = (const float*)d_in[11];
    const float* W1m  = (const float*)d_in[12];
    const float* b1m  = (const float*)d_in[13];
    const float* W1e  = (const float*)d_in[14];
    const float* b1e  = (const float*)d_in[15];
    const float* W2m  = (const float*)d_in[16];
    const float* b2m  = (const float*)d_in[17];
    const float* W2e  = (const float*)d_in[18];
    const float* b2e  = (const float*)d_in[19];
    const float* g1    = (const float*)d_in[20];
    const float* beta1 = (const float*)d_in[21];
    const float* g2    = (const float*)d_in[22];
    const float* beta2 = (const float*)d_in[23];
    float* out = (float*)d_out;

    size_t off = 0;
    auto alloc = [&](size_t bytes) -> void* {
        void* p = (char*)d_ws + off;
        off += (bytes + 255) & ~(size_t)255;
        return p;
    };
    const size_t node_bytes = (size_t)NN * HD * sizeof(float);
    float* X0   = (float*)alloc(node_bytes);          // x_ini
    float* X1   = (float*)alloc(node_bytes);          // h1 / x1
    float* S    = (float*)alloc(node_bytes);          // segment sums; h2/x2 in place
    int*   srcc = (int*)alloc((size_t)E_SEE * 4);
    int*   dstc = (int*)alloc((size_t)E_SEE * 4);
    float* wsum = (float*)alloc((size_t)NN * 4);
    float* degv = (float*)alloc((size_t)NN * 4);
    float* stats = (float*)alloc(2 * HD * 4);

    hipMemsetAsync(S, 0, node_bytes, stream);
    hipMemsetAsync(wsum, 0, (size_t)NN * 4, stream);
    hipMemsetAsync(degv, 0, (size_t)NN * 4, stream);
    hipMemsetAsync(stats, 0, 2 * HD * 4, stream);

    // Input projections -> X0
    gemm_kernel<0><<<NQ / 16, 256, 0, stream>>>(qf, QD, Wq, bq,
        nullptr, nullptr, nullptr, nullptr, nullptr, X0);
    gemm_kernel<0><<<NL / 16, 256, 0, stream>>>(lf, LD, Wl, bl,
        nullptr, nullptr, nullptr, nullptr, nullptr, X0 + (size_t)NQ * HD);

    // Edge compaction + wsum/deg
    edge_prep_kernel<<<(E_SEE + 255) / 256, 256, 0, stream>>>(
        ei, ecansee, eweight, W_em, b_em, srcc, dstc, wsum, degv);

    const unsigned scat_total = (unsigned)E_SEE * HD;
    const unsigned napply     = (unsigned)NN * HD;
    const int rows_pb = 512;
    const int bn_blocks = (NN + rows_pb - 1) / rows_pb;

    // ---- Layer 1 ----
    scatter_kernel<<<(scat_total + 255) / 256, 256, 0, stream>>>(srcc, dstc, X0, S);
    gemm_kernel<1><<<NN / 16, 256, 0, stream>>>(S, HD, W1m, b1m, b1e, W1e,
                                                degv, wsum, X0, X1);
    bn_stats_kernel<<<bn_blocks, 128, 0, stream>>>(X1, stats, rows_pb);
    bn_apply_kernel<<<(napply + 255) / 256, 256, 0, stream>>>(X1, stats, g1, beta1, 1);

    // ---- Layer 2 ----
    hipMemsetAsync(S, 0, node_bytes, stream);
    hipMemsetAsync(stats, 0, 2 * HD * 4, stream);
    scatter_kernel<<<(scat_total + 255) / 256, 256, 0, stream>>>(srcc, dstc, X1, S);
    gemm_kernel<1><<<NN / 16, 256, 0, stream>>>(S, HD, W2m, b2m, b2e, W2e,
                                                degv, wsum, X1, S);   // in place (row-local)
    bn_stats_kernel<<<bn_blocks, 128, 0, stream>>>(S, stats, rows_pb);
    bn_apply_kernel<<<(napply + 255) / 256, 256, 0, stream>>>(S, stats, g2, beta2, 0);

    // ---- Edge prediction ----
    edge_predict_kernel<<<E_PRED / 8, 256, 0, stream>>>(X0, S, ei, emask, out);
}
